// MotionVectorQuantizer_21715354648835
// MI455X (gfx1250) — matrix operand order repaired and checked by reading, not compiled, not run on hardware
//
#include <hip/hip_runtime.h>
#include <hip/hip_bf16.h>
#include <math.h>

// ---------------------------------------------------------------------------
// Fused vector-quantizer for MI455X (gfx1250, wave32, WMMA bf16 + TDM).
//   h = x^T @ pre_w^T + pre_b ; idx = argmax(l2norm(h) @ l2norm(embed)^T)
//   q = en[idx] @ post_w^T + post_b  (straight-through forward)
// - x tile staged HBM -> LDS by the Tensor Data Mover (tensor_load_to_lds)
// - all three GEMMs via v_wmma_f32_16x16x32_bf16, weights L2-resident bf16
// - only HBM-heavy streams: x in (134MB) + q out (134MB)  ->  memory-bound
// ---------------------------------------------------------------------------

typedef __bf16 bf16_t;
typedef __attribute__((ext_vector_type(16))) __bf16 v16bf;
typedef __attribute__((ext_vector_type(8)))  __bf16 v8bf;
typedef __attribute__((ext_vector_type(8)))  float  v8f;
typedef __attribute__((ext_vector_type(4)))  unsigned int v4u;
typedef __attribute__((ext_vector_type(8)))  int v8i;
typedef __attribute__((ext_vector_type(4)))  int v4i;

#define DIM     256
#define KCODES  512
#define BATCH   32
#define SEQ     4096
#define NTOK    (BATCH * SEQ)          // 131072 tokens
#define MTILE   128                    // tokens per block (8 waves x 16 rows)
#define PITCH   264                    // bf16 elements per LDS row (16B-aligned, padded)
#define NBLOCKS (NTOK / MTILE)         // 1024

#define QELEMS  ((size_t)BATCH * DIM * SEQ)   // 33554432
#define IDXBASE (QELEMS + 2)                  // q, commit, perplexity, indices

// workspace layout (bytes)
#define WS_EN      0                                  // 512*256 f32 normalized embed
#define WS_ENB     (WS_EN     + KCODES * DIM * 4)     // 512*256 bf16
#define WS_PREWB   (WS_ENB    + KCODES * DIM * 2)     // 256*256 bf16
#define WS_POSTWB  (WS_PREWB  + DIM * DIM * 2)        // 256*256 bf16
#define WS_COUNTS  (WS_POSTWB + DIM * DIM * 2)        // 512 f32 histogram
#define WS_LOSS    (WS_COUNTS + KCODES * 4)           // 1 f32 commit-loss accum

// LDS layout (xf fp32 staging buffer is dead after phase 1; hb overlays it)
#define XF_BYTES   (MTILE * DIM * 4)                  // 131072: TDM destination
#define XS_BYTES   (MTILE * PITCH * 2)                // 67584
#define SMEM_BYTES (XF_BYTES + XS_BYTES + MTILE * 4 + MTILE * 4 + 256 * 4)

// -------- WMMA fragment loaders (contiguous k order, same in A and B, wave32) ----------------
// A (16x32 bf16): lane m = lane&15; element i -> K = 16*(lane>>4) + i
__device__ __forceinline__ v16bf load_a_frag(const bf16_t* row, int k0, int hi) {
    const v8bf lo = *(const v8bf*)(row + k0 + hi * 16);
    const v8bf hi8 = *(const v8bf*)(row + k0 + hi * 16 + 8);
    return __builtin_shufflevector(lo, hi8, 0,1,2,3,4,5,6,7,8,9,10,11,12,13,14,15);
}
// B (32x16 bf16) with B[k][col] = W[col][k] (row-major W, ldw = DIM):
// lane col = lane&15; element i -> K = 16*(lane>>4) + i  => 16 consecutive bf16
__device__ __forceinline__ v16bf load_b_frag(const bf16_t* W, int col, int k0, int hi) {
    return *(const v16bf*)(W + (size_t)col * DIM + k0 + hi * 16);
}

// ---------------------------- prep kernel -----------------------------------
__global__ void vq_prep(const float* __restrict__ pre_w,
                        const float* __restrict__ post_w,
                        const float* __restrict__ embed,
                        float* __restrict__ en, bf16_t* __restrict__ enb,
                        bf16_t* __restrict__ prewb, bf16_t* __restrict__ postwb,
                        float* __restrict__ counts, float* __restrict__ loss) {
    __shared__ float red[256];
    const int t = threadIdx.x;
    const int r = blockIdx.x;
    if (r == 0) {                       // zero accumulators (stream-ordered)
        counts[t] = 0.0f;
        counts[t + 256] = 0.0f;
        if (t == 0) *loss = 0.0f;
    }
    if (r < KCODES) {                   // normalize one codebook row
        const float v = embed[(size_t)r * DIM + t];
        red[t] = v * v;
        __syncthreads();
        for (int s = 128; s > 0; s >>= 1) {
            if (t < s) red[t] += red[t + s];
            __syncthreads();
        }
        const float inv = 1.0f / sqrtf(red[0] + 1e-12f);
        const float nv = v * inv;
        en[(size_t)r * DIM + t] = nv;
        enb[(size_t)r * DIM + t] = (bf16_t)nv;
    } else if (r < KCODES + DIM) {      // pre_w -> bf16
        const int row = r - KCODES;
        prewb[(size_t)row * DIM + t] = (bf16_t)pre_w[(size_t)row * DIM + t];
    } else {                            // post_w -> bf16
        const int row = r - KCODES - DIM;
        postwb[(size_t)row * DIM + t] = (bf16_t)post_w[(size_t)row * DIM + t];
    }
}

// ---------------------------- main fused kernel ------------------------------
__global__ void __launch_bounds__(256)
vq_main(const float* __restrict__ x,
        const float* __restrict__ pre_b, const float* __restrict__ post_b,
        const float* __restrict__ en, const bf16_t* __restrict__ enb,
        const bf16_t* __restrict__ prewb, const bf16_t* __restrict__ postwb,
        float* __restrict__ counts, float* __restrict__ loss,
        float* __restrict__ out) {
    extern __shared__ char smem[];
    float*  xf      = (float*)smem;                          // TDM dest (phase 1 only)
    bf16_t* hb      = (bf16_t*)smem;                         // h / hn, overlays xf
    bf16_t* xs      = (bf16_t*)(smem + XF_BYTES);            // bf16 A tile (x, then quantize)
    float*  scale_s = (float*)(smem + XF_BYTES + XS_BYTES);  // ||h|| per row
    int*    idx_s   = (int*)(scale_s + MTILE);               // argmax per row
    float*  red     = (float*)(idx_s + MTILE);               // block reduction

    const int tid  = threadIdx.x;
    const int lane = tid & 31;
    const int w    = tid >> 5;          // wave id 0..7, owns rows [w*16, w*16+16)
    const int ln   = lane & 15;
    const int hi   = lane >> 4;
    const int m0   = w * 16;

    const int n0 = blockIdx.x * MTILE;  // first token of tile
    const int bb = n0 >> 12;            // batch index (SEQ = 4096)
    const int l0 = n0 & (SEQ - 1);      // position within sequence

    // ---- phase 1a: TDM DMA of the x tile (fp32) into LDS -------------------
    // 2D tile of tensor x[bb]: dim0 = l (len 4096, contiguous), dim1 = d (256 rows)
    // tile: tile_dim0 = 128 tokens, tile_dim1 = 256 dims; one DMA per block.
    if (w == 0) {
        const unsigned long long ga =
            (unsigned long long)(const void*)(x + ((size_t)bb * DIM) * SEQ + l0);
        const unsigned lds_off = (unsigned)(unsigned long long)(void*)xf;
        v4u g0;
        g0[0] = 1u;                                      // count=1, user descriptor
        g0[1] = lds_off;                                 // lds_addr
        g0[2] = (unsigned)(ga & 0xFFFFFFFFull);          // global_addr[31:0]
        g0[3] = (unsigned)((ga >> 32) & 0x01FFFFFFull)   // global_addr[56:32]
              | (2u << 30);                              // type = 2 ("image")
        v8i g1;
        g1[0] = (int)(2u << 16);                         // data_size = 2 (4 bytes)
        g1[1] = (int)((SEQ & 0xFFFFu) << 16);            // tensor_dim0[15:0] = 4096
        g1[2] = (int)((SEQ >> 16) | ((unsigned)DIM << 16));   // dim0 hi | tensor_dim1 lo
        g1[3] = (int)((DIM >> 16) | ((unsigned)MTILE << 16)); // dim1 hi | tile_dim0 = 128
        g1[4] = (int)DIM;                                // tile_dim1 = 256, tile_dim2 = 0
        g1[5] = (int)SEQ;                                // tensor_dim0_stride[31:0] = 4096
        g1[6] = 0;                                       // stride hi | tensor_dim1_stride lo
        g1[7] = 0;
        const v4i z4 = {0, 0, 0, 0};
#if defined(__clang_major__) && (__clang_major__ >= 23)
        const v8i z8 = {0, 0, 0, 0, 0, 0, 0, 0};
        __builtin_amdgcn_tensor_load_to_lds(g0, g1, z4, z4, z8, 0);
#else
        __builtin_amdgcn_tensor_load_to_lds(g0, g1, z4, z4, 0);
#endif
        __builtin_amdgcn_s_wait_tensorcnt(0);
    }
    __syncthreads();

    // ---- phase 1b: LDS transpose-convert  xf[d][m] (f32) -> xs[m][d] (bf16) -
    for (int e = tid; e < MTILE * DIM; e += 256) {
        const int d = e >> 7;               // 0..255
        const int m = e & (MTILE - 1);      // 0..127  (consecutive tids: no conflicts)
        xs[m * PITCH + d] = (bf16_t)xf[e];
    }
    __syncthreads();

    // ---- phase 2: pre GEMM  h = x_tile @ pre_w^T + pre_b (hb overlays xf) ---
    {
        const bf16_t* arow = xs + (m0 + ln) * PITCH;
        v16bf afrag[8];
#pragma unroll
        for (int ks = 0; ks < 8; ++ks) afrag[ks] = load_a_frag(arow, ks * 32, hi);
#pragma unroll 2
        for (int it = 0; it < 16; ++it) {
            v8f acc = {};
#pragma unroll
            for (int ks = 0; ks < 8; ++ks) {
                const v16bf b = load_b_frag(prewb, it * 16 + ln, ks * 32, hi);
                acc = __builtin_amdgcn_wmma_f32_16x16x32_bf16(
                    false, afrag[ks], false, b, (short)0, acc, false, false);
            }
            const float bias = pre_b[it * 16 + ln];
#pragma unroll
            for (int r = 0; r < 8; ++r)
                hb[(m0 + hi * 8 + r) * PITCH + it * 16 + ln] = (bf16_t)(acc[r] + bias);
        }
    }
    __syncthreads();

    // ---- phase 3: row L2-normalize h in place (hn), keep ||h|| -------------
    if (tid < MTILE) {
        bf16_t* row = hb + tid * PITCH;
        float s = 0.0f;
        for (int k = 0; k < DIM; ++k) { const float v = (float)row[k]; s += v * v; }
        const float nrm = sqrtf(s + 1e-12f);
        scale_s[tid] = nrm;
        const float inv = 1.0f / nrm;
        for (int k = 0; k < DIM; ++k) row[k] = (bf16_t)((float)row[k] * inv);
    }
    __syncthreads();

    // ---- phase 4: sim GEMM hn @ en^T with fused row-argmax -----------------
    {
        const bf16_t* arow = hb + (m0 + ln) * PITCH;
        v16bf afrag[8];
#pragma unroll
        for (int ks = 0; ks < 8; ++ks) afrag[ks] = load_a_frag(arow, ks * 32, hi);
        float best[8];
        int   bidx[8];
#pragma unroll
        for (int r = 0; r < 8; ++r) { best[r] = -1e30f; bidx[r] = 0; }
        for (int jt = 0; jt < KCODES / 16; ++jt) {      // 32 code tiles
            v8f acc = {};
#pragma unroll
            for (int ks = 0; ks < 8; ++ks) {
                const v16bf b = load_b_frag(enb, jt * 16 + ln, ks * 32, hi);
                acc = __builtin_amdgcn_wmma_f32_16x16x32_bf16(
                    false, afrag[ks], false, b, (short)0, acc, false, false);
            }
            const int col = jt * 16 + ln;
#pragma unroll
            for (int r = 0; r < 8; ++r) {
                if (acc[r] > best[r]) { best[r] = acc[r]; bidx[r] = col; }
            }
        }
        // reduce (max, first-index) across the 16 lanes sharing each row half
#pragma unroll
        for (int off = 1; off < 16; off <<= 1) {
#pragma unroll
            for (int r = 0; r < 8; ++r) {
                const float ov = __shfl_xor(best[r], off, 32);
                const int   oi = __shfl_xor(bidx[r], off, 32);
                if (ov > best[r] || (ov == best[r] && oi < bidx[r])) {
                    best[r] = ov; bidx[r] = oi;
                }
            }
        }
        if (ln == 0) {
#pragma unroll
            for (int r = 0; r < 8; ++r) idx_s[m0 + hi * 8 + r] = bidx[r];
        }
    }
    __syncthreads();

    // ---- phase 5: histogram, indices out, commit loss, stage quantized -----
    if (tid < MTILE) {
        const int id = idx_s[tid];
        atomicAdd(counts + id, 1.0f);
        out[IDXBASE + (size_t)n0 + tid] = (float)id;
    }
    float local = 0.0f;
    for (int e = tid; e < MTILE * DIM; e += 256) {
        const int m = e >> 8;
        const int k = e & (DIM - 1);
        const int id = idx_s[m];
        const float hv = (float)hb[m * PITCH + k] * scale_s[m];   // un-normalized h
        const float qv = en[(size_t)id * DIM + k];
        const float d = qv - hv;
        local += d * d;
        xs[m * PITCH + k] = enb[(size_t)id * DIM + k];            // stage quantize
    }
    red[tid] = local;
    __syncthreads();
    for (int s = 128; s > 0; s >>= 1) {
        if (tid < s) red[tid] += red[tid + s];
        __syncthreads();
    }
    if (tid == 0) atomicAdd(loss, red[0]);

    // ---- phase 6: post GEMM  q = quantize @ post_w^T + post_b, store (B,D,L)
    {
        const bf16_t* arow = xs + (m0 + ln) * PITCH;
        v16bf afrag[8];
#pragma unroll
        for (int ks = 0; ks < 8; ++ks) afrag[ks] = load_a_frag(arow, ks * 32, hi);
#pragma unroll 2
        for (int it = 0; it < 16; ++it) {
            v8f acc = {};
#pragma unroll
            for (int ks = 0; ks < 8; ++ks) {
                const v16bf b = load_b_frag(postwb, it * 16 + ln, ks * 32, hi);
                acc = __builtin_amdgcn_wmma_f32_16x16x32_bf16(
                    false, afrag[ks], false, b, (short)0, acc, false, false);
            }
            const int col = it * 16 + ln;
            const float pb = post_b[col];
            // lane holds 8 consecutive tokens (along L) at fixed output dim col
            float* p = out + (((size_t)bb * DIM + col) << 12) + l0 + m0 + hi * 8;
            *(float4*)p       = make_float4(acc[0] + pb, acc[1] + pb, acc[2] + pb, acc[3] + pb);
            *(float4*)(p + 4) = make_float4(acc[4] + pb, acc[5] + pb, acc[6] + pb, acc[7] + pb);
        }
    }
}

// ---------------------------- finalize kernel --------------------------------
__global__ void vq_finalize(const float* __restrict__ counts,
                            const float* __restrict__ loss,
                            float* __restrict__ out) {
    __shared__ float red[KCODES];
    const int t = threadIdx.x;
    const float p = counts[t] / (float)NTOK;
    red[t] = -p * logf(p + 1e-10f);
    __syncthreads();
    for (int s = KCODES / 2; s > 0; s >>= 1) {
        if (t < s) red[t] += red[t + s];
        __syncthreads();
    }
    if (t == 0) {
        out[QELEMS]     = 0.25f * (*loss) / (float)((size_t)NTOK * DIM);  // commit
        out[QELEMS + 1] = expf(red[0]);                                    // perplexity
    }
}

// ---------------------------- launch ----------------------------------------
extern "C" void kernel_launch(void* const* d_in, const int* in_sizes, int n_in,
                              void* d_out, int out_size, void* d_ws, size_t ws_size,
                              hipStream_t stream) {
    const float* x      = (const float*)d_in[0];
    const float* pre_w  = (const float*)d_in[1];
    const float* pre_b  = (const float*)d_in[2];
    const float* post_w = (const float*)d_in[3];
    const float* post_b = (const float*)d_in[4];
    const float* embed  = (const float*)d_in[5];
    float* out = (float*)d_out;

    char* ws = (char*)d_ws;
    float*  en     = (float*)(ws + WS_EN);
    bf16_t* enb    = (bf16_t*)(ws + WS_ENB);
    bf16_t* prewb  = (bf16_t*)(ws + WS_PREWB);
    bf16_t* postwb = (bf16_t*)(ws + WS_POSTWB);
    float*  counts = (float*)(ws + WS_COUNTS);
    float*  loss   = (float*)(ws + WS_LOSS);

    vq_prep<<<KCODES + 2 * DIM, 256, 0, stream>>>(pre_w, post_w, embed,
                                                  en, enb, prewb, postwb,
                                                  counts, loss);

    // ~196 KB dynamic LDS: legal on CDNA5 (320 KB per workgroup/WGP)
    (void)hipFuncSetAttribute((const void*)vq_main,
                              hipFuncAttributeMaxDynamicSharedMemorySize,
                              (int)SMEM_BYTES);
    vq_main<<<NBLOCKS, 256, SMEM_BYTES, stream>>>(x, pre_b, post_b,
                                                  en, enb, prewb, postwb,
                                                  counts, loss, out);

    vq_finalize<<<1, KCODES, 0, stream>>>(counts, loss, out);
}
